// EDESC_18296560681209
// MI455X (gfx1250) — matrix operand order repaired and checked by reading, not compiled, not run on hardware
//
#include <hip/hip_runtime.h>

// ---------------------------------------------------------------------------
// s = rownorm(((sumsq_per_512block(z @ D)) + ETA*512) / (6*512)),  out = (s, z)
//
// Pipeline:
//   1) z  (f32) -> bf16 staging        (in d_out's z region, written last)
//   2) D  (f32) -> bf16 transposed Dt  (so B tiles are per-lane contiguous)
//   3) GEMM: TDM (tensor_load_to_lds) double-buffered staging + bf16 WMMA,
//      zD never materialized; per-cluster sum-of-squares folded via atomics
//   4) memcpy z into d_out, then finalize (affine + row-normalize)
// ---------------------------------------------------------------------------

typedef __attribute__((ext_vector_type(16))) __bf16 v16bf;
typedef __attribute__((ext_vector_type(8)))  float  v8f;
typedef unsigned uint32x4 __attribute__((ext_vector_type(4)));
typedef int      int32x8  __attribute__((ext_vector_type(8)));
typedef int      int32x4  __attribute__((ext_vector_type(4)));

#define N_ROWS     16384
#define N_ZDIM     8192
#define N_CLUSTERS 16
#define D_SUB      512
#define ETA        5.0f

#define BM 128
#define BN 128
#define BK 32

// LDS tile layout: per row/col, 32 bf16 = 16 dwords data + 1 dword pad
// (TDM pad_interval=16 dwords, pad_amount=1 dword reproduces this stride).
#define ROW_U    17                       // uints per row incl. pad
#define TILE_U   (BM * ROW_U)             // 2176 uints per A or B tile
#define TILE_B   (TILE_U * 4)             // 8704 bytes
#define BUF_U    (2 * TILE_U)             // A tile + B tile per buffer
#define BUF_B    (BUF_U * 4)              // 17408 bytes

union FragBF { v16bf v; unsigned u[8]; };

static __device__ __forceinline__ unsigned pack_bf16(float lo, float hi) {
    __bf16 l = (__bf16)lo, h = (__bf16)hi;
    unsigned short ul, uh;
    __builtin_memcpy(&ul, &l, 2);
    __builtin_memcpy(&uh, &h, 2);
    return (unsigned)ul | ((unsigned)uh << 16);
}

// ---------------------------------------------------------------------------
// TDM: issue a 2D tile load (tile_dim0 = 32 bf16 along contiguous axis,
// tile_dim1 = 128 rows, row stride 8192 elements) into LDS at lds_byte_off.
// D# bit layout per CDNA5 ISA ch.8 (groups 0/1; groups 2/3 unused for 2D).
// ---------------------------------------------------------------------------
static __device__ __forceinline__ void tdm_load_tile(
    const unsigned short* gsrc, unsigned lds_byte_off, unsigned tensor_dim1) {
    unsigned long long a = (unsigned long long)(size_t)gsrc;
    uint32x4 g0;
    g0[0] = 1u;                                        // count=1, user desc
    g0[1] = lds_byte_off;                              // lds_addr
    g0[2] = (unsigned)(a & 0xFFFFFFFFu);               // global_addr[31:0]
    g0[3] = ((unsigned)(a >> 32) & 0x01FFFFFFu)        // global_addr[56:32]
            | 0x80000000u;                             // type=2 ("image")
    int32x8 g1;
    // wg_mask=0 | data_size=1(2B)<<16 | pad_enable<<20 | pad_interval=3(16dw)<<22
    // | pad_amount=0(1dw)<<25
    g1[0] = (int)0x00D10000u;
    g1[1] = (int)(( (unsigned)N_ZDIM & 0xFFFFu) << 16);   // tensor_dim0 lo16 @48
    g1[2] = (int)((tensor_dim1 & 0xFFFFu) << 16);         // dim0 hi=0 | dim1 lo16
    g1[3] = (int)((unsigned)BK << 16);                    // dim1 hi=0 | tile_dim0=32
    g1[4] = (int)BM;                                      // tile_dim1=128, tile_dim2=0
    g1[5] = (int)N_ZDIM;                                  // tensor_dim0_stride lo32
    g1[6] = 0;                                            // stride hi | dim1_stride lo
    g1[7] = 0;
    int32x4 gz4 = {0, 0, 0, 0};
    int32x8 gz8 = {0, 0, 0, 0, 0, 0, 0, 0};
    // amdgpu-toolchain (clang-23) 6-arg form
    __builtin_amdgcn_tensor_load_to_lds(g0, g1, gz4, gz4, gz8, 0);
}

// ---------------------------------------------------------------------------
// Main kernel: 128x128 zD tile per block, TDM double-buffered bf16 staging,
// 8 WMMAs/wave/K-step, sum-of-squares epilogue into s[row][cluster].
// ---------------------------------------------------------------------------
__global__ __launch_bounds__(256) void gemm_sumsq_tdm_kernel(
    const unsigned short* __restrict__ zt,   // [16384][8192] bf16 row-major
    const unsigned short* __restrict__ Dt,   // [8192 cols][8192 k] bf16 (D^T)
    float* __restrict__ s_acc) {

    // Single LDS object => starts at LDS offset 0 (descriptor assumption).
    __shared__ unsigned lds[2 * BUF_U];      // 34816 bytes

    const int tid  = threadIdx.x;
    const int lane = tid & 31;
    const int wave = tid >> 5;
    const int wm   = wave & 3;      // 4 waves along M, 32 rows each
    const int wn   = wave >> 2;     // 2 waves along N, 64 cols each
    const int half = lane >> 4;
    const int l16  = lane & 15;

    const int rowBase = blockIdx.y * BM;
    const int colBase = blockIdx.x * BN;
    const int cluster = colBase / D_SUB;

    v8f acc[2][4] = {};

    // Prologue: stage buffer 0 (wave0 -> A tile, wave1 -> B tile)
    if (wave == 0)
        tdm_load_tile(zt + (size_t)rowBase * N_ZDIM, 0u, N_ROWS);
    else if (wave == 1)
        tdm_load_tile(Dt + (size_t)colBase * N_ZDIM, (unsigned)TILE_B, N_ZDIM);

    const int NT = N_ZDIM / BK;     // 256 K-steps
    for (int t = 0; t < NT; ++t) {
        // All waves: waves that issued nothing have TENSORcnt==0 (free).
        __builtin_amdgcn_s_wait_tensorcnt(0);   // buf[t&1] landed in LDS
        __syncthreads();                        // release all waves; also
                                                // compute(t-1) done => safe to
                                                // overwrite buf[(t+1)&1]

        // Compiler barrier: TDM writes LDS out-of-band; make `lds` a
        // may-write target so fragment loads are re-issued every iteration.
        asm volatile("" : : "v"((const void*)lds) : "memory");

        if (t + 1 < NT) {
            const int kn = (t + 1) * BK;
            const unsigned nb = ((t + 1) & 1) ? (unsigned)BUF_B : 0u;
            if (wave == 0)
                tdm_load_tile(zt + (size_t)rowBase * N_ZDIM + kn, nb, N_ROWS);
            else if (wave == 1)
                tdm_load_tile(Dt + (size_t)colBase * N_ZDIM + kn, nb + TILE_B,
                              N_ZDIM);
        }

        const unsigned Abase = (t & 1) ? (unsigned)BUF_U : 0u;
        const unsigned Bbase = Abase + TILE_U;

        // ---- gather fragments (ISA 7.12.2 16-bit layouts) ------------------
        // A 16x32: lanes 0-15/16-31 hold rows M=0..15;
        //   uint v holds K = (v<4?0:16) + half*8 + (v&3)*2
        FragBF a[2], b[4];
        #pragma unroll
        for (int mt = 0; mt < 2; ++mt) {
            const int row = wm * 32 + mt * 16 + l16;
            #pragma unroll
            for (int v = 0; v < 8; ++v) {
                const int ui = row * ROW_U + ((v < 4) ? 0 : 8) + half * 4 + (v & 3);
                a[mt].u[v] = lds[Abase + ui];
            }
        }
        // B 32x16: N=l16; same K order as A:
        //   uint v holds K = (v<4?0:16) + half*8 + (v&3)*2
        #pragma unroll
        for (int nt = 0; nt < 4; ++nt) {
            const int col = wn * 64 + nt * 16 + l16;
            #pragma unroll
            for (int v = 0; v < 8; ++v)
                b[nt].u[v] = lds[Bbase + col * ROW_U + ((v < 4) ? 0 : 8) + half * 4 + (v & 3)];
        }

        #pragma unroll
        for (int mt = 0; mt < 2; ++mt)
            #pragma unroll
            for (int nt = 0; nt < 4; ++nt)
                acc[mt][nt] = __builtin_amdgcn_wmma_f32_16x16x32_bf16(
                    false, a[mt].v, false, b[nt].v,
                    (short)0, acc[mt][nt], false, false);
    }

    // ---- epilogue: sum of squares over this tile's 128 columns -------------
    // C/D layout: lanes 0-15 -> M = i, lanes 16-31 -> M = 8 + i; N = l16.
    #pragma unroll
    for (int mt = 0; mt < 2; ++mt) {
        #pragma unroll
        for (int i = 0; i < 8; ++i) {
            float p = 0.f;
            #pragma unroll
            for (int nt = 0; nt < 4; ++nt) {
                const float v = acc[mt][nt][i];
                p += v * v;
            }
            #pragma unroll
            for (int m = 1; m < 16; m <<= 1)   // masks <16 stay in wave32 half
                p += __shfl_xor(p, m, 32);
            if (l16 == 0) {
                const int row = rowBase + wm * 32 + mt * 16 + half * 8 + i;
                atomicAdd(&s_acc[row * N_CLUSTERS + cluster], p);
            }
        }
    }
}

// ---------------------------------------------------------------------------
// f32 -> packed bf16 (straight copy, for z)
// ---------------------------------------------------------------------------
__global__ __launch_bounds__(256) void convert_bf16_kernel(
    const float* __restrict__ in, unsigned* __restrict__ out, int n4) {
    const int i = blockIdx.x * blockDim.x + threadIdx.x;
    if (i < n4) {
        const float4 f = ((const float4*)in)[i];
        uint2 p;
        p.x = pack_bf16(f.x, f.y);
        p.y = pack_bf16(f.z, f.w);
        ((uint2*)out)[i] = p;
    }
}

// ---------------------------------------------------------------------------
// f32 -> bf16 with transpose (for D -> Dt[col][k]); 32x32 tiles via LDS
// ---------------------------------------------------------------------------
__global__ __launch_bounds__(256) void transpose_bf16_kernel(
    const float* __restrict__ D, unsigned short* __restrict__ Dt) {
    __shared__ float tile[32][33];
    const int tx = threadIdx.x, ty = threadIdx.y;     // blockDim (32, 8)
    const int x  = blockIdx.x * 32 + tx;              // col index in D
    const int y0 = blockIdx.y * 32;                   // k base
    #pragma unroll
    for (int j = 0; j < 32; j += 8)
        tile[ty + j][tx] = D[(size_t)(y0 + ty + j) * N_ZDIM + x];
    __syncthreads();
    const int xo = y0 + tx;                           // k index (contiguous out)
    const int c0 = blockIdx.x * 32;                   // col base
    #pragma unroll
    for (int j = 0; j < 32; j += 8) {
        const __bf16 h = (__bf16)tile[tx][ty + j];
        unsigned short u;
        __builtin_memcpy(&u, &h, 2);
        Dt[(size_t)(c0 + ty + j) * N_ZDIM + xo] = u;
    }
}

__global__ void zero_kernel(float* __restrict__ p, int n) {
    const int i = blockIdx.x * blockDim.x + threadIdx.x;
    if (i < n) p[i] = 0.f;
}

__global__ void finalize_kernel(float* __restrict__ s) {
    const int row = blockIdx.x * blockDim.x + threadIdx.x;
    if (row >= N_ROWS) return;
    const float scale = 1.0f / ((ETA + 1.0f) * (float)D_SUB);
    float v[N_CLUSTERS];
    float sum = 0.f;
    #pragma unroll
    for (int c = 0; c < N_CLUSTERS; ++c) {
        const float t = (s[row * N_CLUSTERS + c] + ETA * (float)D_SUB) * scale;
        v[c] = t;
        sum += t;
    }
    const float inv = 1.0f / sum;
    #pragma unroll
    for (int c = 0; c < N_CLUSTERS; ++c)
        s[row * N_CLUSTERS + c] = v[c] * inv;
}

extern "C" void kernel_launch(void* const* d_in, const int* in_sizes, int n_in,
                              void* d_out, int out_size, void* d_ws, size_t ws_size,
                              hipStream_t stream) {
    const float* z = (const float*)d_in[0];   // [16384, 8192] f32
    const float* D = (const float*)d_in[1];   // [8192, 8192]  f32
    float* out   = (float*)d_out;
    float* s     = out;                                   // [16384, 16]
    float* z_out = out + (size_t)N_ROWS * N_CLUSTERS;     // [16384, 8192]

    // Stage bf16 operands inside the z-output region (overwritten at the end):
    //   zt: 256 MB, Dt: 128 MB  (region is 512 MB)
    unsigned short* zt = (unsigned short*)z_out;
    unsigned short* Dt = zt + (size_t)N_ROWS * N_ZDIM;

    // 1) zero the s accumulator region
    zero_kernel<<<(N_ROWS * N_CLUSTERS + 255) / 256, 256, 0, stream>>>(
        s, N_ROWS * N_CLUSTERS);

    // 2) z -> bf16
    {
        const int n4 = (int)(((size_t)N_ROWS * N_ZDIM) / 4);
        convert_bf16_kernel<<<(n4 + 255) / 256, 256, 0, stream>>>(
            z, (unsigned*)zt, n4);
    }
    // 3) D -> bf16 transposed
    {
        dim3 blk(32, 8);
        dim3 grd(N_ZDIM / 32, N_ZDIM / 32);
        transpose_bf16_kernel<<<grd, blk, 0, stream>>>(D, Dt);
    }

    // 4) fused GEMM (TDM-staged, bf16 WMMA) + per-cluster sum-of-squares
    {
        dim3 grid(N_ZDIM / BN, N_ROWS / BM);   // (64, 128)
        gemm_sumsq_tdm_kernel<<<grid, 256, 0, stream>>>(zt, Dt, s);
    }

    // 5) second output is z itself (overwrites staging, stream-ordered)
    (void)hipMemcpyAsync(z_out, z, (size_t)N_ROWS * N_ZDIM * sizeof(float),
                         hipMemcpyDeviceToDevice, stream);

    // 6) affine + row-normalize in place
    finalize_kernel<<<(N_ROWS + 255) / 256, 256, 0, stream>>>(s);
}
